// LocalPointModel_67242007986287
// MI455X (gfx1250) — matrix operand order repaired and checked by reading, not compiled, not run on hardware
//
#include <hip/hip_runtime.h>

#define BB 2
#define NN 8192
#define KNEI 16
#define HH 128
#define GG 256
#define NCLS 10
#define NPRM 8
#define BN (BB*NN)

typedef __attribute__((ext_vector_type(16))) __bf16 v16bf;
typedef __attribute__((ext_vector_type(8)))  __bf16 v8bf;
typedef __attribute__((ext_vector_type(8)))  float  v8f;

__device__ __forceinline__ v8f wmma_bf16(v16bf a, v16bf b, v8f c) {
  return __builtin_amdgcn_wmma_f32_16x16x32_bf16(false, a, false, b, (short)0, c,
                                                 false, false);
}

// CDNA5 async global->LDS copy (16 bytes per lane), tracked by ASYNCcnt.
__device__ __forceinline__ void async_copy_b128(unsigned lds_off, const void* g) {
  asm volatile("global_load_async_to_lds_b128 %0, %1, off"
               : : "v"(lds_off), "v"(g) : "memory");
}
__device__ __forceinline__ void wait_async() {
  asm volatile("s_wait_asynccnt 0x0" : : : "memory");
}

// A-fragment (16x32 bf16): lane = half*16+m holds row m.
// element e -> K = kc*32 + ((e<8)?e:e+8) + half*8  (two contiguous 8-element runs)
__device__ __forceinline__ v16bf ld_afrag(const __bf16* row, int kc, int half) {
  v8bf lo = *(const v8bf*)(row + kc * 32 + half * 8);
  v8bf hi = *(const v8bf*)(row + kc * 32 + 16 + half * 8);
  return __builtin_shufflevector(lo, hi, 0,1,2,3,4,5,6,7,8,9,10,11,12,13,14,15);
}

// B-fragment (32x16 bf16) from transposed weights Wt[n][Kdim]:
// lane holds column n = lane&15; element e -> K = kc*32 + ((e<8)?e:e+8) + half*8
__device__ __forceinline__ v16bf ld_bfrag(const __bf16* wt, int Kdim, int n,
                                          int kc, int half) {
  const __bf16* r = wt + (size_t)n * Kdim + kc * 32 + half * 8;
  v8bf lo = *(const v8bf*)(r);
  v8bf hi = *(const v8bf*)(r + 16);
  return __builtin_shufflevector(lo, hi, 0,1,2,3,4,5,6,7,8,9,10,11,12,13,14,15);
}

// ---------------- weight convert: fp32 [K][N] -> bf16 transposed [Npad][K] ----
__global__ void cvt_t_kernel(const float* __restrict__ src, __bf16* __restrict__ dst,
                             int kdim, int ndim, int npad) {
  int i = blockIdx.x * 256 + threadIdx.x;
  if (i >= kdim * npad) return;
  int n = i / kdim, k = i % kdim;
  dst[i] = (n < ndim) ? (__bf16)src[(size_t)k * ndim + n] : (__bf16)0.f;
}

__global__ void init_gf_kernel(int* gfi) {
  int i = blockIdx.x * 256 + threadIdx.x;
  if (i < BB * GG) gfi[i] = 0;  // 0.0f
}

// ---------------- per-point MLP: pf = relu(relu(x@We1+be1)@We2+be2) ----------
__global__ void __launch_bounds__(128)
pf_kernel(const float* __restrict__ points, const float* __restrict__ normals,
          const float* __restrict__ We1, const float* __restrict__ be1,
          const float* __restrict__ We2, const float* __restrict__ be2,
          float* __restrict__ pf, __bf16* __restrict__ pfb) {
  int p = blockIdx.x;          // global point 0..BN
  int c = threadIdx.x;         // output channel
  float xin[6];
#pragma unroll
  for (int i = 0; i < 3; ++i) {
    xin[i]     = points[(size_t)p * 3 + i];
    xin[3 + i] = normals[(size_t)p * 3 + i];
  }
  float h = be1[c];
#pragma unroll
  for (int i = 0; i < 6; ++i) h += xin[i] * We1[i * HH + c];
  h = fmaxf(h, 0.f);
  __shared__ float h1[HH];
  h1[c] = h;
  __syncthreads();
  float h2 = be2[c];
  for (int i = 0; i < HH; ++i) h2 += h1[i] * We2[i * HH + c];
  h2 = fmaxf(h2, 0.f);
  pf[(size_t)p * HH + c]  = h2;
  pfb[(size_t)p * HH + c] = (__bf16)h2;
}

// ---------------- KNN: one wave per query, LDS-staged points ------------------
__global__ void __launch_bounds__(256)
knn_kernel(const float* __restrict__ points, int* __restrict__ knn) {
  extern __shared__ float spts[];               // 4096*3 floats = 48KB
  int wave = threadIdx.x >> 5, lane = threadIdx.x & 31;
  int b = blockIdx.x / (NN / 8);
  int q = (blockIdx.x % (NN / 8)) * 8 + wave;   // query within batch
  const float* P = points + (size_t)b * NN * 3;
  float qx = P[q * 3 + 0], qy = P[q * 3 + 1], qz = P[q * 3 + 2];

  float d[17]; int id[17];
#pragma unroll
  for (int s = 0; s < 17; ++s) { d[s] = 3.0e38f; id[s] = 0; }

  for (int ch = 0; ch < 2; ++ch) {
    __syncthreads();
    for (int i = threadIdx.x; i < 4096 * 3; i += 256)
      spts[i] = P[ch * 4096 * 3 + i];
    __syncthreads();
    for (int t = lane; t < 4096; t += 32) {
      float dx = spts[t * 3 + 0] - qx;
      float dy = spts[t * 3 + 1] - qy;
      float dz = spts[t * 3 + 2] - qz;
      float d2 = dx * dx + dy * dy + dz * dz;
      if (d2 < d[16]) {
        d[16] = d2; id[16] = ch * 4096 + t;
#pragma unroll
        for (int s = 16; s > 0; --s) {
          if (d[s] < d[s - 1]) {
            float td = d[s]; d[s] = d[s - 1]; d[s - 1] = td;
            int ti = id[s]; id[s] = id[s - 1]; id[s - 1] = ti;
          }
        }
      }
    }
  }
  // merge 17 smallest across the wave; round 0 discards the self point
  int myidx = 0;
#pragma unroll 1
  for (int r = 0; r < 17; ++r) {
    float v = d[0]; int pid = id[0]; int owner = lane;
#pragma unroll
    for (int off = 16; off >= 1; off >>= 1) {
      float ov = __shfl_xor(v, off, 32);
      int   op = __shfl_xor(pid, off, 32);
      int   oo = __shfl_xor(owner, off, 32);
      if (ov < v || (ov == v && op < pid)) { v = ov; pid = op; owner = oo; }
    }
    if (lane == owner) {
#pragma unroll
      for (int s = 0; s < 16; ++s) { d[s] = d[s + 1]; id[s] = id[s + 1]; }
      d[16] = 3.0e38f;
    }
    if (r > 0 && lane == r - 1) myidx = pid;
  }
  if (lane < 16) knn[((size_t)b * NN + q) * KNEI + lane] = myidx;
}

// ---------------- edge MLP + max over K: one wave per point -------------------
// LDS: Wl1t (64KB) | Wl2t (32KB) | 8 x edge slab (8KB) = 160KB total
__global__ void __launch_bounds__(256)
edge_kernel(const float* __restrict__ pf, const int* __restrict__ knn,
            const __bf16* __restrict__ Wl12t,  // Wl1t followed by Wl2t (contiguous)
            const float* __restrict__ bl1, const float* __restrict__ bl2,
            __bf16* __restrict__ lfb) {
  extern __shared__ __bf16 smem_e[];
  __bf16* wl1s = smem_e;                        // 32768 elems
  __bf16* wl2s = smem_e + 32768;                // 16384 elems
  int wave = threadIdx.x >> 5, lane = threadIdx.x & 31;
  int half = lane >> 4, m = lane & 15;
  __bf16* eds = smem_e + 49152 + wave * 16 * 256;
  int p = blockIdx.x * 8 + wave;                // global point
  int b = p >> 13;

  // async-stage both weight matrices (96KB) into LDS
  {
    unsigned lbase = (unsigned)(size_t)(void*)smem_e;
    for (int i = threadIdx.x * 16; i < 96 * 1024; i += 256 * 16)
      async_copy_b128(lbase + i, (const char*)Wl12t + i);
  }

  // stage edge matrix: row j = [ctr(128), nbr_j - ctr(128)]
  float ctr[4];
#pragma unroll
  for (int c4 = 0; c4 < 4; ++c4) ctr[c4] = pf[(size_t)p * HH + lane + 32 * c4];
#pragma unroll 1
  for (int j = 0; j < KNEI; ++j) {
    int g = (b << 13) + knn[(size_t)p * KNEI + j];
#pragma unroll
    for (int c4 = 0; c4 < 4; ++c4) {
      int c = lane + 32 * c4;
      float f = pf[(size_t)g * HH + c];
      eds[j * 256 + c]       = (__bf16)ctr[c4];
      eds[j * 256 + 128 + c] = (__bf16)(f - ctr[c4]);
    }
  }
  wait_async();
  __syncthreads();

  v16bf a[8];
#pragma unroll
  for (int kc = 0; kc < 8; ++kc) a[kc] = ld_afrag(eds + m * 256, kc, half);
  __syncthreads();

  __bf16* h1 = eds;                             // reuse LDS slab (16x128)
  int col = m;
#pragma unroll 1
  for (int nt = 0; nt < 8; nt += 2) {           // 2 tiles: independent WMMA chains
    v8f acc0 = {}, acc1 = {};
#pragma unroll
    for (int kc = 0; kc < 8; ++kc) {
      v16bf b0 = ld_bfrag(wl1s, 256, nt * 16 + col, kc, half);
      v16bf b1 = ld_bfrag(wl1s, 256, (nt + 1) * 16 + col, kc, half);
      acc0 = wmma_bf16(a[kc], b0, acc0);
      acc1 = wmma_bf16(a[kc], b1, acc1);
    }
    int nc0 = nt * 16 + col, nc1 = (nt + 1) * 16 + col;
    float bi0 = bl1[nc0], bi1 = bl1[nc1];
#pragma unroll
    for (int e = 0; e < 8; ++e) {
      h1[(e + half * 8) * HH + nc0] = (__bf16)fmaxf(acc0[e] + bi0, 0.f);
      h1[(e + half * 8) * HH + nc1] = (__bf16)fmaxf(acc1[e] + bi1, 0.f);
    }
  }
  __syncthreads();

  v16bf a2[4];
#pragma unroll
  for (int kc = 0; kc < 4; ++kc) a2[kc] = ld_afrag(h1 + m * HH, kc, half);
#pragma unroll 1
  for (int nt = 0; nt < 8; nt += 2) {
    v8f acc0 = {}, acc1 = {};
#pragma unroll
    for (int kc = 0; kc < 4; ++kc) {
      v16bf b0 = ld_bfrag(wl2s, 128, nt * 16 + col, kc, half);
      v16bf b1 = ld_bfrag(wl2s, 128, (nt + 1) * 16 + col, kc, half);
      acc0 = wmma_bf16(a2[kc], b0, acc0);
      acc1 = wmma_bf16(a2[kc], b1, acc1);
    }
#pragma unroll
    for (int u = 0; u < 2; ++u) {
      int nc = (nt + u) * 16 + col;
      float bias = bl2[nc];
      float vmax = 0.f;
#pragma unroll
      for (int e = 0; e < 8; ++e) {
        float x = (u ? acc1[e] : acc0[e]) + bias;
        vmax = fmaxf(vmax, fmaxf(x, 0.f));
      }
      vmax = fmaxf(vmax, __shfl_xor(vmax, 16, 32));    // max over all 16 rows
      if (half == 0) lfb[(size_t)p * HH + nc] = (__bf16)vmax;
    }
  }
}

// ---------------- ctx + global max-pool: one wave per 16-point tile ----------
// LDS: Wct staged (128KB) via async copies
__global__ void __launch_bounds__(256)
ctx_kernel(const __bf16* __restrict__ pfb, const __bf16* __restrict__ lfb,
           const __bf16* __restrict__ Wct, const float* __restrict__ bc,
           int* __restrict__ gfi) {
  extern __shared__ __bf16 smem_c[];            // 65536 elems = 128KB
  int wave = threadIdx.x >> 5, lane = threadIdx.x & 31;
  int half = lane >> 4, m = lane & 15;
  int t = blockIdx.x * 8 + wave;
  int p0 = t * 16, b = p0 >> 13;
  int g = p0 + m;

  {
    unsigned lbase = (unsigned)(size_t)(void*)smem_c;
    for (int i = threadIdx.x * 16; i < 128 * 1024; i += 256 * 16)
      async_copy_b128(lbase + i, (const char*)Wct + i);
  }

  v16bf a[8];
#pragma unroll
  for (int kc = 0; kc < 4; ++kc) a[kc]     = ld_afrag(pfb + (size_t)g * HH, kc, half);
#pragma unroll
  for (int kc = 0; kc < 4; ++kc) a[kc + 4] = ld_afrag(lfb + (size_t)g * HH, kc, half);

  wait_async();
  __syncthreads();

#pragma unroll 1
  for (int nt = 0; nt < 16; nt += 2) {
    v8f acc0 = {}, acc1 = {};
#pragma unroll
    for (int kc = 0; kc < 8; ++kc) {
      v16bf b0 = ld_bfrag(smem_c, 256, nt * 16 + m, kc, half);
      v16bf b1 = ld_bfrag(smem_c, 256, (nt + 1) * 16 + m, kc, half);
      acc0 = wmma_bf16(a[kc], b0, acc0);
      acc1 = wmma_bf16(a[kc], b1, acc1);
    }
#pragma unroll
    for (int u = 0; u < 2; ++u) {
      int nc = (nt + u) * 16 + m;
      float bias = bc[nc];
      float vmax = 0.f;
#pragma unroll
      for (int e = 0; e < 8; ++e) {
        float x = (u ? acc1[e] : acc0[e]) + bias;
        vmax = fmaxf(vmax, fmaxf(x, 0.f));
      }
      vmax = fmaxf(vmax, __shfl_xor(vmax, 16, 32));
      if (half == 0) atomicMax(&gfi[b * GG + nc], __float_as_int(vmax));
    }
  }
}

// ---------------- heads: fused(512) -> 128 -> {10, 8} ------------------------
__global__ void __launch_bounds__(256)
head_kernel(const __bf16* __restrict__ pfb, const __bf16* __restrict__ lfb,
            const float* __restrict__ gf,
            const __bf16* __restrict__ Wk1t, const float* __restrict__ bk1,
            const __bf16* __restrict__ Wk2t, const float* __restrict__ bk2,
            const __bf16* __restrict__ Wp1t, const float* __restrict__ bp1,
            const __bf16* __restrict__ Wp2t, const float* __restrict__ bp2,
            float* __restrict__ out) {
  __shared__ __bf16 hbuf[8][16 * HH];
  int wave = threadIdx.x >> 5, lane = threadIdx.x & 31;
  int half = lane >> 4, m = lane & 15;
  int t = blockIdx.x * 8 + wave;
  int p0 = t * 16, b = p0 >> 13;
  int g = p0 + m;

  v16bf a[16];
#pragma unroll
  for (int kc = 0; kc < 4; ++kc) a[kc]     = ld_afrag(pfb + (size_t)g * HH, kc, half);
#pragma unroll
  for (int kc = 0; kc < 4; ++kc) a[kc + 4] = ld_afrag(lfb + (size_t)g * HH, kc, half);
#pragma unroll
  for (int kc = 8; kc < 16; ++kc) {
    v16bf tv;
#pragma unroll
    for (int e = 0; e < 16; ++e) {
      int ko = ((e < 8) ? e : (e + 8)) + half * 8;
      tv[e] = (__bf16)gf[b * GG + (kc - 8) * 32 + ko];
    }
    a[kc] = tv;
  }

  __bf16* h1 = hbuf[wave];
  float* outL = out;
  float* outP = out + (size_t)BN * NCLS;

#pragma unroll 1
  for (int hd = 0; hd < 2; ++hd) {
    const __bf16* W1 = hd ? Wp1t : Wk1t;  const float* b1 = hd ? bp1 : bk1;
    const __bf16* W2 = hd ? Wp2t : Wk2t;  const float* b2 = hd ? bp2 : bk2;
    int oc = hd ? NPRM : NCLS;

#pragma unroll 1
    for (int nt = 0; nt < 8; nt += 2) {
      v8f acc0 = {}, acc1 = {};
#pragma unroll
      for (int kc = 0; kc < 16; ++kc) {
        v16bf b0 = ld_bfrag(W1, 512, nt * 16 + m, kc, half);
        v16bf b1f = ld_bfrag(W1, 512, (nt + 1) * 16 + m, kc, half);
        acc0 = wmma_bf16(a[kc], b0, acc0);
        acc1 = wmma_bf16(a[kc], b1f, acc1);
      }
      int nc0 = nt * 16 + m, nc1 = (nt + 1) * 16 + m;
      float bi0 = b1[nc0], bi1 = b1[nc1];
#pragma unroll
      for (int e = 0; e < 8; ++e) {
        h1[(e + half * 8) * HH + nc0] = (__bf16)fmaxf(acc0[e] + bi0, 0.f);
        h1[(e + half * 8) * HH + nc1] = (__bf16)fmaxf(acc1[e] + bi1, 0.f);
      }
    }
    __syncthreads();

    v16bf a2[4];
#pragma unroll
    for (int kc = 0; kc < 4; ++kc) a2[kc] = ld_afrag(h1 + m * HH, kc, half);
    v8f acc = {};
#pragma unroll
    for (int kc = 0; kc < 4; ++kc)
      acc = wmma_bf16(a2[kc], ld_bfrag(W2, 128, m, kc, half), acc);
    float bias = (m < oc) ? b2[m] : 0.f;
    float* dst = hd ? outP : outL;
#pragma unroll
    for (int e = 0; e < 8; ++e) {
      int row = e + half * 8;
      if (m < oc) dst[(size_t)(p0 + row) * oc + m] = acc[e] + bias;
    }
    __syncthreads();
  }
}

// =============================================================================
extern "C" void kernel_launch(void* const* d_in, const int* in_sizes, int n_in,
                              void* d_out, int out_size, void* d_ws, size_t ws_size,
                              hipStream_t stream) {
  const float* points  = (const float*)d_in[0];
  const float* normals = (const float*)d_in[1];
  const float* We1 = (const float*)d_in[2];  const float* be1 = (const float*)d_in[3];
  const float* We2 = (const float*)d_in[4];  const float* be2 = (const float*)d_in[5];
  const float* Wl1 = (const float*)d_in[6];  const float* bl1 = (const float*)d_in[7];
  const float* Wl2 = (const float*)d_in[8];  const float* bl2 = (const float*)d_in[9];
  const float* Wc  = (const float*)d_in[10]; const float* bc  = (const float*)d_in[11];
  const float* Wk1 = (const float*)d_in[12]; const float* bk1 = (const float*)d_in[13];
  const float* Wk2 = (const float*)d_in[14]; const float* bk2 = (const float*)d_in[15];
  const float* Wp1 = (const float*)d_in[16]; const float* bp1 = (const float*)d_in[17];
  const float* Wp2 = (const float*)d_in[18]; const float* bp2 = (const float*)d_in[19];

  char* ws = (char*)d_ws;
  size_t o = 0;
  float*  pf   = (float*)(ws + o);  o += (size_t)BN * HH * 4;   // 8 MB
  __bf16* pfb  = (__bf16*)(ws + o); o += (size_t)BN * HH * 2;   // 4 MB
  __bf16* lfb  = (__bf16*)(ws + o); o += (size_t)BN * HH * 2;   // 4 MB
  int*    knn  = (int*)(ws + o);    o += (size_t)BN * KNEI * 4; // 1 MB
  float*  gf   = (float*)(ws + o);  o += (size_t)BB * GG * 4;
  __bf16* Wl1t = (__bf16*)(ws + o); o += 128 * 256 * 2;   // contiguous with Wl2t!
  __bf16* Wl2t = (__bf16*)(ws + o); o += 128 * 128 * 2;
  __bf16* Wct  = (__bf16*)(ws + o); o += 256 * 256 * 2;
  __bf16* Wk1t = (__bf16*)(ws + o); o += 128 * 512 * 2;
  __bf16* Wk2t = (__bf16*)(ws + o); o += 16 * 128 * 2;
  __bf16* Wp1t = (__bf16*)(ws + o); o += 128 * 512 * 2;
  __bf16* Wp2t = (__bf16*)(ws + o); o += 16 * 128 * 2;

  // weight conversions (fp32 row-major [K][N] -> bf16 transposed [Npad][K])
  cvt_t_kernel<<<(256 * 128 + 255) / 256, 256, 0, stream>>>(Wl1, Wl1t, 256, 128, 128);
  cvt_t_kernel<<<(128 * 128 + 255) / 256, 256, 0, stream>>>(Wl2, Wl2t, 128, 128, 128);
  cvt_t_kernel<<<(256 * 256 + 255) / 256, 256, 0, stream>>>(Wc,  Wct,  256, 256, 256);
  cvt_t_kernel<<<(512 * 128 + 255) / 256, 256, 0, stream>>>(Wk1, Wk1t, 512, 128, 128);
  cvt_t_kernel<<<(128 * 16  + 255) / 256, 256, 0, stream>>>(Wk2, Wk2t, 128, 10,  16);
  cvt_t_kernel<<<(512 * 128 + 255) / 256, 256, 0, stream>>>(Wp1, Wp1t, 512, 128, 128);
  cvt_t_kernel<<<(128 * 16  + 255) / 256, 256, 0, stream>>>(Wp2, Wp2t, 128, 8,   16);
  init_gf_kernel<<<2, 256, 0, stream>>>((int*)gf);

  pf_kernel<<<BN, 128, 0, stream>>>(points, normals, We1, be1, We2, be2, pf, pfb);
  knn_kernel<<<BB * (NN / 8), 256, 4096 * 3 * sizeof(float), stream>>>(points, knn);
  edge_kernel<<<BN / 8, 256, (96 * 1024) + 8 * 16 * 256 * sizeof(__bf16), stream>>>(
      pf, knn, Wl1t, bl1, bl2, lfb);
  ctx_kernel<<<BN / 16 / 8, 256, 128 * 1024, stream>>>(pfb, lfb, Wct, bc, (int*)gf);
  head_kernel<<<BN / 16 / 8, 256, 0, stream>>>(pfb, lfb, gf, Wk1t, bk1, Wk2t, bk2,
                                               Wp1t, bp1, Wp2t, bp2, (float*)d_out);
}
